// UnifiedConnectionClassifier_55731495632923
// MI455X (gfx1250) — matrix operand order repaired and checked by reading, not compiled, not run on hardware
//
#include <hip/hip_runtime.h>
#include <cmath>

typedef __attribute__((ext_vector_type(16))) _Float16 v16h;
typedef __attribute__((ext_vector_type(8)))  float    v8f;

#define BATCH 32768
#define NK    26
#define SSZ   32
#define HID   64
#define NPAIR (BATCH * NK)   /* 851968 */
#define NTILE (NPAIR / 16)   /* 53248, exact */

// Branchless tanh via hardware exp2/rcp: tanh(x) = 1 - 2/(exp2(2*log2e*x)+1).
// Saturates to +/-1 at large |x| with no NaN path, no EXEC divergence.
__device__ __forceinline__ float fast_tanh(float x) {
  float t = __builtin_amdgcn_exp2f(x * 2.8853900817779268f);  // exp(2x)
  float r = __builtin_amdgcn_rcpf(t + 1.0f);
  return __builtin_fmaf(-2.0f, r, 1.0f);
}

// sigmoid(x) = 1/(1 + exp(-x)) via exp2/rcp.
__device__ __forceinline__ float fast_sigmoid(float x) {
  float t = __builtin_amdgcn_exp2f(x * -1.4426950408889634f); // exp(-x)
  return __builtin_amdgcn_rcpf(t + 1.0f);
}

__global__ __launch_bounds__(256) void ucc_classify_kernel(
    const int*   __restrict__ cellIdx,   // [BATCH]
    const int*   __restrict__ nbIdx,     // [BATCH*NK]
    const float* __restrict__ states,    // [262144, 32]
    const float* __restrict__ W1,        // [64, 64]  (k, n)
    const float* __restrict__ b1,        // [64]
    const float* __restrict__ W2,        // [64, 1]
    const float* __restrict__ b2,        // [1]
    const float* __restrict__ thr,       // [4]
    float*       __restrict__ out)       // [5 * NPAIR]
{
  const int lane  = threadIdx.x & 31;
  const int nBase = lane & 15;     // N column within 16-wide tile / tile row id
  const int half  = lane >> 4;     // lane half selects K sub-range per ISA layout
  const int waveId = (blockIdx.x * blockDim.x + threadIdx.x) >> 5;
  const int nWaves = (gridDim.x * blockDim.x) >> 5;

  // ---- One-time: W1 as f16 B-fragments (32x16 each): 2 K-chunks x 4 N-tiles.
  // B layout (16-bit, 32x16): lanes 0-15 hold K {0..7, 16..23}, lanes 16-31 hold K {8..15, 24..31},
  // N = lane&15, matching the A-fragment K order.
  v16h bfr[2][4];
  {
    const int kb = half * 8;
    #pragma unroll
    for (int kc = 0; kc < 2; ++kc)
      #pragma unroll
      for (int nt = 0; nt < 4; ++nt) {
        v16h f;
        #pragma unroll
        for (int i = 0; i < 16; ++i)
          f[i] = (_Float16)W1[(kc * 32 + kb + i + (i >= 8 ? 8 : 0)) * HID + nt * 16 + nBase];
        bfr[kc][nt] = f;
      }
  }
  float bias1[4], w2r[4];
  #pragma unroll
  for (int nt = 0; nt < 4; ++nt) {
    bias1[nt] = b1[nt * 16 + nBase];
    w2r[nt]   = W2[nt * 16 + nBase];
  }
  const float bias2  = b2[0];
  const float tLocal = thr[0], tFunc = thr[1], tDist = thr[2], tSim = thr[3];
  const int grp = half * 8;   // A-fragment K group offset for this lane half

  float* outSims = out;
  float* outLoc  = out + (size_t)NPAIR;
  float* outFun  = out + (size_t)NPAIR * 2;
  float* outDis  = out + (size_t)NPAIR * 3;
  float* outVal  = out + (size_t)NPAIR * 4;

  for (int tile = waveId; tile < NTILE; tile += nWaves) {
    // Lane's A-row = pair index p; lanes l and l+16 share a row (different K cols).
    const int p  = tile * 16 + nBase;
    const int ci = cellIdx[p / NK];
    int ni = nbIdx[p];
    ni = ni < 0 ? 0 : ni;                       // clamp invalid (same as reference)
    const float* cs = states + (size_t)ci * SSZ;
    const float* ns = states + (size_t)ni * SSZ;

    // A-fragments (16-bit A 16x32): lane half 0 -> K {0..7, 16..23},
    // half 1 -> K {8..15, 24..31}; chunk 0 = cell state, chunk 1 = neighbor state.
    v16h afr[2];
    #pragma unroll
    for (int kc = 0; kc < 2; ++kc) {
      const float* src = kc ? ns : cs;
      const float4 q0 = *(const float4*)(src + grp);
      const float4 q1 = *(const float4*)(src + grp + 4);
      const float4 q2 = *(const float4*)(src + 16 + grp);
      const float4 q3 = *(const float4*)(src + 16 + grp + 4);
      v16h a;
      a[0]  = (_Float16)q0.x; a[1]  = (_Float16)q0.y; a[2]  = (_Float16)q0.z; a[3]  = (_Float16)q0.w;
      a[4]  = (_Float16)q1.x; a[5]  = (_Float16)q1.y; a[6]  = (_Float16)q1.z; a[7]  = (_Float16)q1.w;
      a[8]  = (_Float16)q2.x; a[9]  = (_Float16)q2.y; a[10] = (_Float16)q2.z; a[11] = (_Float16)q2.w;
      a[12] = (_Float16)q3.x; a[13] = (_Float16)q3.y; a[14] = (_Float16)q3.z; a[15] = (_Float16)q3.w;
      afr[kc] = a;
    }

    // Layer 1 GEMM: h[16x64] = pair[16x64] @ W1[64x64], f32 accumulate.
    v8f acc[4] = {};
    #pragma unroll
    for (int nt = 0; nt < 4; ++nt) {
      acc[nt] = __builtin_amdgcn_wmma_f32_16x16x32_f16(
          false, afr[0], false, bfr[0][nt], (short)0, acc[nt], false, false);
      acc[nt] = __builtin_amdgcn_wmma_f32_16x16x32_f16(
          false, afr[1], false, bfr[1][nt], (short)0, acc[nt], false, false);
    }

    // Layer 2: per-lane partial over its N column, for rows m = v + 8*half.
    // Branchless tanh keeps EXEC full and co-issues with the matrix pipe.
    float partial[8];
    #pragma unroll
    for (int v = 0; v < 8; ++v) {
      float s = 0.0f;
      #pragma unroll
      for (int nt = 0; nt < 4; ++nt)
        s = __builtin_fmaf(fast_tanh(acc[nt][v] + bias1[nt]), w2r[nt], s);
      partial[v] = s;
    }
    // Reduce across the 16 lanes of each half (N dimension).
    #pragma unroll
    for (int v = 0; v < 8; ++v) {
      #pragma unroll
      for (int off = 1; off < 16; off <<= 1)
        partial[v] += __shfl_xor(partial[v], off, 32);
    }

    // Writers: lanes 0..7 emit rows 0..7, lanes 16..23 emit rows 8..15.
    if (nBase < 8) {
      const int v = nBase;
      float s = partial[0];
      #pragma unroll
      for (int i = 1; i < 8; ++i)
        if (v == i) s = partial[i];        // cndmask chain, no scratch
      const int m  = v + 8 * half;
      const int p2 = tile * 16 + m;
      const float sim = fast_sigmoid(s + bias2);

      const int c2   = cellIdx[p2 / NK];
      const int nraw = nbIdx[p2];
      const bool valid = nraw >= 0;
      const int n2 = valid ? nraw : 0;
      const float dx = (float)(c2 >> 12)       - (float)(n2 >> 12);
      const float dy = (float)((c2 >> 6) & 63) - (float)((n2 >> 6) & 63);
      const float dz = (float)(c2 & 63)        - (float)(n2 & 63);
      const float d  = sqrtf(dx * dx + dy * dy + dz * dz);

      const bool loc = (d <= tLocal) && valid;
      const bool fc  = (d > tLocal) && (d <= tFunc);
      const bool mid = (d > tFunc) && (d < tDist);
      const bool fun = (fc || (mid && (sim > tSim))) && valid;
      const bool dis = (d >= tDist) && valid;

      outSims[p2] = sim;
      outLoc[p2]  = loc   ? 1.0f : 0.0f;
      outFun[p2]  = fun   ? 1.0f : 0.0f;
      outDis[p2]  = dis   ? 1.0f : 0.0f;
      outVal[p2]  = valid ? 1.0f : 0.0f;
    }
  }
}

extern "C" void kernel_launch(void* const* d_in, const int* in_sizes, int n_in,
                              void* d_out, int out_size, void* d_ws, size_t ws_size,
                              hipStream_t stream) {
  (void)in_sizes; (void)n_in; (void)out_size; (void)d_ws; (void)ws_size;
  const int*   cellIdx = (const int*)  d_in[0];
  const int*   nbIdx   = (const int*)  d_in[1];
  const float* states  = (const float*)d_in[2];
  const float* W1      = (const float*)d_in[3];
  const float* b1      = (const float*)d_in[4];
  const float* W2      = (const float*)d_in[5];
  const float* b2      = (const float*)d_in[6];
  const float* thr     = (const float*)d_in[7];
  float* out = (float*)d_out;

  // 1024 blocks x 8 waves = 8192 waves; ~6.5 tiles each (amortizes the
  // one-time W1 fragment load), plenty of occupancy for the WGP array.
  dim3 grid(1024), block(256);
  ucc_classify_kernel<<<grid, block, 0, stream>>>(
      cellIdx, nbIdx, states, W1, b1, W2, b2, thr, out);
}
